// NeuralCDE_62740882260323
// MI455X (gfx1250) — matrix operand order repaired and checked by reading, not compiled, not run on hardware
//
#include <hip/hip_runtime.h>
#include <hip/hip_bf16.h>

typedef __attribute__((ext_vector_type(16))) __bf16 v16bf;
typedef __attribute__((ext_vector_type(8)))  __bf16 v8bf;
typedef __attribute__((ext_vector_type(4)))  __bf16 v4bf;
typedef __attribute__((ext_vector_type(8)))  float  v8f;

// ---- constants (problem sizes from reference) ----
// B=1024, L=200, D=32, H=64, W=128, OUT=16, N_STEPS=100
#define ACT_NONE 0
#define ACT_RELU 1
#define ACT_SP   2
#define MODE_BF16     0   // bias+act -> bf16 LDS (stride 128)
#define MODE_F32      1   // bias+act -> f32 LDS (stride 64)
#define MODE_CONTRACT 2   // (c+bias)*dx, lane-reduce, ds_add into k (stride 64)
#define MODE_OUT      3   // bias -> f32 global (stride 16)

// packed bf16 weight-fragment offsets in d_ws (elements)
static constexpr size_t W_IW1 = 0;        // 128x32  -> 4096
static constexpr size_t W_IW2 = 4096;     // 128x128 -> 16384
static constexpr size_t W_IW3 = 20480;    // 64x128  -> 8192
static constexpr size_t W_FW1 = 28672;    // 128x64  -> 8192
static constexpr size_t W_FW2 = 36864;    // 128x128 -> 16384
static constexpr size_t W_FW3 = 53248;    // 2048x128-> 262144
static constexpr size_t W_DW1 = 315392;   // 128x64  -> 8192
static constexpr size_t W_DW2 = 323584;   // 128x128 -> 16384
static constexpr size_t W_DW3 = 339968;   // 16x128  -> 2048
// total 342016 bf16 = 684 KB workspace

template<int ACT>
__device__ __forceinline__ float activate(float v) {
  if (ACT == ACT_RELU) return fmaxf(v, 0.f);
  if (ACT == ACT_SP)   return fmaxf(v, 0.f) + __logf(1.f + __expf(-fabsf(v)));
  return v;
}

__device__ __forceinline__ void store4bf(__bf16* dst, float a0, float a1, float a2, float a3) {
  v4bf p = { (__bf16)a0, (__bf16)a1, (__bf16)a2, (__bf16)a3 };
  *(v4bf*)dst = p;   // 8-byte ds_store_b64
}

// ---------------- pre-pass: pack f32 weight [N,K] into bf16 WMMA B-fragments ----------------
// fragment order: frag = nt*(K/32)+kb ; within frag: 32 lanes x 16 bf16 (32B/lane)
// lane<16 : b[j] = W[nt*16+lane][kb*32 + j]       (K rows 0..15 of block)
// lane>=16: b[j] = W[nt*16+lane-16][kb*32+16+j]   (K rows 16..31 of block)
__global__ void pack_weights(const float* __restrict__ Wsrc, __bf16* __restrict__ dst,
                             int N, int K) {
  int kblocks = K >> 5;
  int total = N * K;
  int e = blockIdx.x * blockDim.x + threadIdx.x;
  if (e >= total) return;
  int frag = e >> 9;
  int r = e & 511;
  int lane = r >> 4;
  int j = r & 15;
  int nt = frag / kblocks;
  int kb = frag - nt * kblocks;
  int n = nt * 16 + (lane & 15);
  int k = kb * 32 + ((lane >> 4) << 4) + j;
  dst[e] = (__bf16)(Wsrc[(size_t)n * K + k]);
}

// ---------------- generic M=16 WMMA GEMM stage ----------------
template<int KB, int ACT, int MODE>
__device__ __forceinline__ void gemm_stage(
    const __bf16* Asrc,            // LDS activations, row stride 128
    int ntiles,
    const __bf16* __restrict__ Bg, // global packed fragments
    const float* biasL,            // LDS bias
    void* dst,
    const float* dxL,              // LDS dx (CONTRACT only)
    int lane, int wave)
{
  const int half = lane >> 4;
  const int nidx = lane & 15;
  v16bf a[KB];
#pragma unroll
  for (int kb = 0; kb < KB; ++kb) {
    const __bf16* p = Asrc + nidx * 128 + kb * 32 + half * 16;
    v8bf lo = *(const v8bf*)(p);
    v8bf hi = *(const v8bf*)(p + 8);
    a[kb] = __builtin_shufflevector(lo, hi, 0,1,2,3,4,5,6,7,8,9,10,11,12,13,14,15);
  }

  // CONTRACT: d = (nt&1)*16+nidx, and nt strides by 8 from wave, so nt&1 == wave&1
  // -> the 8 dx values per lane are loop-invariant; hoist out of the tile loop.
  float dxv[8];
  if (MODE == MODE_CONTRACT) {
    const int dd = ((wave & 1) << 4) + nidx;
#pragma unroll
    for (int r = 0; r < 8; ++r)
      dxv[r] = dxL[(half * 8 + r) * 32 + dd];
  }

#pragma unroll 2
  for (int nt = wave; nt < ntiles; nt += 8) {
    v8f c = {};
    const __bf16* bp = Bg + ((size_t)nt * KB) * 512 + lane * 16;
#pragma unroll
    for (int kb = 0; kb < KB; ++kb) {
      v16bf b = *(const v16bf*)(bp + kb * 512);
      c = __builtin_amdgcn_wmma_f32_16x16x32_bf16(false, a[kb], false, b,
                                                  (short)0, c, false, false);
    }
    if (MODE == MODE_BF16) {
      const int n0 = nt * 16;
      const float bias = biasL[n0 + nidx];
      __bf16* d = (__bf16*)dst;
#pragma unroll
      for (int r = 0; r < 8; ++r)
        d[(half * 8 + r) * 128 + n0 + nidx] = (__bf16)(activate<ACT>(c[r] + bias));
    } else if (MODE == MODE_F32) {
      const int n0 = nt * 16;
      const float bias = biasL[n0 + nidx];
      float* d = (float*)dst;
#pragma unroll
      for (int r = 0; r < 8; ++r)
        d[(half * 8 + r) * 64 + n0 + nidx] = activate<ACT>(c[r] + bias);
    } else if (MODE == MODE_CONTRACT) {
      // tile nt covers hd = nt*16 + nidx ; h = nt>>1
      const int h = nt >> 1;
      const float bias = biasL[nt * 16 + nidx];
      float t[8];
#pragma unroll
      for (int r = 0; r < 8; ++r)
        t[r] = (c[r] + bias) * dxv[r];
#pragma unroll
      for (int m = 1; m < 16; m <<= 1) {
#pragma unroll
        for (int r = 0; r < 8; ++r) t[r] += __shfl_xor(t[r], m, 32);
      }
      if (nidx == 0) {
        float* kd = (float*)dst;
#pragma unroll
        for (int r = 0; r < 8; ++r)
          atomicAdd(&kd[(half * 8 + r) * 64 + h], t[r]);
      }
    } else { // MODE_OUT
      const float bias = biasL[nidx];
      float* o = (float*)dst;
#pragma unroll
      for (int r = 0; r < 8; ++r)
        o[(size_t)(half * 8 + r) * 16 + nidx] = activate<ACT>(c[r] + bias);
    }
  }
}

// evaluate spline derivative coefficients for one time point into LDS [16x32]
// vectorized: 128 threads each handle one float4 (full 128B coalesced segments)
__device__ __forceinline__ void compute_dx(
    float t, float t0,
    const float* __restrict__ cb, const float* __restrict__ cc,
    const float* __restrict__ cd, int bbase, float* dxL, int tid)
{
  float rel = t - t0;                       // ts = t0 + j (arange)
  int i = (int)floorf(rel);
  i = i < 0 ? 0 : (i > 198 ? 198 : i);
  float frac = rel - (float)i;
  if (tid < 128) {
    int b = tid >> 3, d4 = (tid & 7) << 2;
    size_t off = ((size_t)(bbase + b) * 199 + i) * 32 + d4;
    float4 vb = *(const float4*)(cb + off);
    float4 vc = *(const float4*)(cc + off);
    float4 vd = *(const float4*)(cd + off);
    float4 r;
    r.x = vb.x + frac * (2.f * vc.x + 3.f * frac * vd.x);
    r.y = vb.y + frac * (2.f * vc.y + 3.f * frac * vd.y);
    r.z = vb.z + frac * (2.f * vc.z + 3.f * frac * vd.z);
    r.w = vb.w + frac * (2.f * vc.w + 3.f * frac * vd.w);
    *(float4*)(dxL + (size_t)tid * 4) = r;
  }
}

__device__ __forceinline__ void vf_eval(
    __bf16* Aa, __bf16* Ab, float* kdst,
    const __bf16* __restrict__ wbf, const float* biasL, const float* dxL,
    int lane, int wave)
{
  gemm_stage<2, ACT_SP, MODE_BF16>(Aa, 8, wbf + W_FW1, biasL + 320, Ab, nullptr, lane, wave);
  __syncthreads();
  gemm_stage<4, ACT_SP, MODE_BF16>(Ab, 8, wbf + W_FW2, biasL + 448, Aa, nullptr, lane, wave);
  __syncthreads();
  gemm_stage<4, ACT_NONE, MODE_CONTRACT>(Aa, 128, wbf + W_FW3, biasL + 576, kdst, dxL, lane, wave);
  __syncthreads();
}

// ---------------- persistent kernel: one WG per 16-row batch tile ----------------
__global__ void __launch_bounds__(256)
ncde_kernel(const float* __restrict__ ts,
            const float* __restrict__ coef_d, const float* __restrict__ coef_c,
            const float* __restrict__ coef_b, const float* __restrict__ coef_a,
            const float* __restrict__ ib1, const float* __restrict__ ib2,
            const float* __restrict__ ib3,
            const float* __restrict__ fb1, const float* __restrict__ fb2,
            const float* __restrict__ fb3,
            const float* __restrict__ db1, const float* __restrict__ db2,
            const float* __restrict__ db3,
            const __bf16* __restrict__ wbf,
            float* __restrict__ out)
{
  __shared__ __attribute__((aligned(32))) __bf16 Aa[16 * 128];
  __shared__ __attribute__((aligned(32))) __bf16 Ab[16 * 128];
  __shared__ __attribute__((aligned(16))) float yf[16 * 64];
  __shared__ __attribute__((aligned(16))) float k1[16 * 64];
  __shared__ __attribute__((aligned(16))) float k2[16 * 64];
  __shared__ __attribute__((aligned(16))) float dx1[16 * 32];
  __shared__ __attribute__((aligned(16))) float dx2[16 * 32];
  __shared__ float biasL[2896];
  // bias layout: ib1@0 ib2@128 ib3@256 fb1@320 fb2@448 fb3@576 db1@2624 db2@2752 db3@2880

  const int tid = threadIdx.x;
  const int lane = tid & 31, wave = tid >> 5;
  const int bbase = blockIdx.x * 16;

  for (int i = tid; i < 128; i += 256) {
    biasL[i]        = ib1[i];
    biasL[128 + i]  = ib2[i];
    biasL[320 + i]  = fb1[i];
    biasL[448 + i]  = fb2[i];
    biasL[2624 + i] = db1[i];
    biasL[2752 + i] = db2[i];
  }
  for (int i = tid; i < 64; i += 256)   biasL[256 + i]  = ib3[i];
  for (int i = tid; i < 2048; i += 256) biasL[576 + i]  = fb3[i];
  for (int i = tid; i < 16; i += 256)   biasL[2880 + i] = db3[i];

  // x = coef_a[:, 0, :] -> bf16 A   (float4 loads, packed b64 LDS stores)
  if (tid < 128) {
    int b = tid >> 3, d4 = (tid & 7) << 2;
    float4 v = *(const float4*)(coef_a + ((size_t)(bbase + b) * 199) * 32 + d4);
    store4bf(&Aa[b * 128 + d4], v.x, v.y, v.z, v.w);
  }
  const float t0 = ts[0];
  const float dt = (ts[199] - t0) * 0.01f;
  __syncthreads();

  // ---- initial MLP: y0 = sp(sp(sp(x@iW1+ib1)@iW2+ib2)@iW3+ib3) ----
  gemm_stage<1, ACT_SP, MODE_BF16>(Aa, 8, wbf + W_IW1, biasL + 0,   Ab, nullptr, lane, wave);
  __syncthreads();
  gemm_stage<4, ACT_SP, MODE_BF16>(Ab, 8, wbf + W_IW2, biasL + 128, Aa, nullptr, lane, wave);
  __syncthreads();
  gemm_stage<4, ACT_SP, MODE_F32 >(Aa, 4, wbf + W_IW3, biasL + 256, yf, nullptr, lane, wave);
  __syncthreads();
  {
    int b = tid >> 4, h4 = (tid & 15) << 2;     // 256 threads x float4 = 1024
    float4 v = *(const float4*)(yf + b * 64 + h4);
    store4bf(&Aa[b * 128 + h4], v.x, v.y, v.z, v.w);
  }

  // ---- 100 Heun steps ----
  for (int step = 0; step < 100; ++step) {
    float tcur = t0 + (float)step * dt;

    // phase A: zero k1/k2, evaluate BOTH dx vectors for this step
    // (global-load latency for eval2's coefficients overlaps eval1's)
    {
      float4 z = {0.f, 0.f, 0.f, 0.f};
      ((float4*)k1)[tid] = z;
      ((float4*)k2)[tid] = z;
    }
    compute_dx(tcur,      t0, coef_b, coef_c, coef_d, bbase, dx1, tid);
    compute_dx(tcur + dt, t0, coef_b, coef_c, coef_d, bbase, dx2, tid);
    __syncthreads();

    vf_eval(Aa, Ab, k1, wbf, biasL, dx1, lane, wave);     // k1 = vf(t, y)

    {                                                     // Aa = bf16(y + dt*k1)
      int b = tid >> 4, h4 = (tid & 15) << 2;
      float4 yv = *(const float4*)(yf + b * 64 + h4);
      float4 kv = *(const float4*)(k1 + b * 64 + h4);
      store4bf(&Aa[b * 128 + h4], yv.x + dt * kv.x, yv.y + dt * kv.y,
                                  yv.z + dt * kv.z, yv.w + dt * kv.w);
    }
    __syncthreads();

    vf_eval(Aa, Ab, k2, wbf, biasL, dx2, lane, wave);     // k2 = vf(t+dt, y+dt*k1)

    {                                                     // y += dt/2*(k1+k2)
      int b = tid >> 4, h4 = (tid & 15) << 2;
      float4 yv = *(const float4*)(yf + b * 64 + h4);
      float4 k1v = *(const float4*)(k1 + b * 64 + h4);
      float4 k2v = *(const float4*)(k2 + b * 64 + h4);
      float4 v;
      v.x = yv.x + 0.5f * dt * (k1v.x + k2v.x);
      v.y = yv.y + 0.5f * dt * (k1v.y + k2v.y);
      v.z = yv.z + 0.5f * dt * (k1v.z + k2v.z);
      v.w = yv.w + 0.5f * dt * (k1v.w + k2v.w);
      *(float4*)(yf + b * 64 + h4) = v;
      store4bf(&Aa[b * 128 + h4], v.x, v.y, v.z, v.w);
    }
    __syncthreads();
  }

  // ---- decoder: Linear->relu->Linear->relu->Linear ----
  gemm_stage<2, ACT_RELU, MODE_BF16>(Aa, 8, wbf + W_DW1, biasL + 2624, Ab, nullptr, lane, wave);
  __syncthreads();
  gemm_stage<4, ACT_RELU, MODE_BF16>(Ab, 8, wbf + W_DW2, biasL + 2752, Aa, nullptr, lane, wave);
  __syncthreads();
  gemm_stage<4, ACT_NONE, MODE_OUT>(Aa, 1, wbf + W_DW3, biasL + 2880,
                                    out + (size_t)bbase * 16, nullptr, lane, wave);
}

extern "C" void kernel_launch(void* const* d_in, const int* in_sizes, int n_in,
                              void* d_out, int out_size, void* d_ws, size_t ws_size,
                              hipStream_t stream) {
  (void)in_sizes; (void)n_in; (void)out_size; (void)ws_size;
  const float* ts     = (const float*)d_in[0];
  const float* coef_d = (const float*)d_in[1];
  const float* coef_c = (const float*)d_in[2];
  const float* coef_b = (const float*)d_in[3];
  const float* coef_a = (const float*)d_in[4];
  const float* iW1 = (const float*)d_in[5];  const float* ib1 = (const float*)d_in[6];
  const float* iW2 = (const float*)d_in[7];  const float* ib2 = (const float*)d_in[8];
  const float* iW3 = (const float*)d_in[9];  const float* ib3 = (const float*)d_in[10];
  const float* fW1 = (const float*)d_in[11]; const float* fb1 = (const float*)d_in[12];
  const float* fW2 = (const float*)d_in[13]; const float* fb2 = (const float*)d_in[14];
  const float* fW3 = (const float*)d_in[15]; const float* fb3 = (const float*)d_in[16];
  const float* dW1 = (const float*)d_in[17]; const float* db1 = (const float*)d_in[18];
  const float* dW2 = (const float*)d_in[19]; const float* db2 = (const float*)d_in[20];
  const float* dW3 = (const float*)d_in[21]; const float* db3 = (const float*)d_in[22];

  __bf16* wbf = (__bf16*)d_ws;

  struct Pack { const float* W; int N; int K; size_t off; };
  const Pack packs[9] = {
      {iW1, 128, 32,  W_IW1}, {iW2, 128, 128, W_IW2}, {iW3, 64,  128, W_IW3},
      {fW1, 128, 64,  W_FW1}, {fW2, 128, 128, W_FW2}, {fW3, 2048,128, W_FW3},
      {dW1, 128, 64,  W_DW1}, {dW2, 128, 128, W_DW2}, {dW3, 16,  128, W_DW3}};
  for (int i = 0; i < 9; ++i) {
    int total = packs[i].N * packs[i].K;
    int blocks = (total + 255) / 256;
    pack_weights<<<blocks, 256, 0, stream>>>(packs[i].W, wbf + packs[i].off,
                                             packs[i].N, packs[i].K);
  }

  ncde_kernel<<<64, 256, 0, stream>>>(ts, coef_d, coef_c, coef_b, coef_a,
                                      ib1, ib2, ib3, fb1, fb2, fb3,
                                      db1, db2, db3, wbf, (float*)d_out);
}
